// GNNItemTower_37306085933638
// MI455X (gfx1250) — matrix operand order repaired and checked by reading, not compiled, not run on hardware
//
#include <hip/hip_runtime.h>

// ---------------------------------------------------------------------------
// 2-layer GCN (GCNConv x2) for MI455X / gfx1250, wave32 + WMMA bf16.
//   h1  = x @ W1                      (WMMA f32_16x16x32_bf16)
//   a1  = relu(scatter(norm * h1[src]) + b1)
//   h2  = a1 @ W2                     (WMMA)
//   out = scatter(norm * h2[src]) + b2
// norm = dinv[src]*dinv[dst], dinv = rsqrt(deg), deg from dst histogram (+self loop)
// ---------------------------------------------------------------------------

typedef __bf16   v16bf __attribute__((ext_vector_type(16)));
typedef float    v8f   __attribute__((ext_vector_type(8)));

#define GCN_N     100000
#define GCN_DIN   512
#define GCN_DH    128
#define GCN_DOUT  64

// -------------------------------- degree / norm ----------------------------

__global__ void k_init_deg(float* __restrict__ deg, int n) {
    int i = blockIdx.x * blockDim.x + threadIdx.x;
    if (i < n) deg[i] = 1.0f;                 // self-loop contribution
}

__global__ void k_deg_hist(const int* __restrict__ dst, float* __restrict__ deg, int e) {
    int i = blockIdx.x * blockDim.x + threadIdx.x;
    if (i < e) atomicAdd(&deg[dst[i]], 1.0f);
}

__global__ void k_deg_to_dinv(float* __restrict__ deg, int n) {
    int i = blockIdx.x * blockDim.x + threadIdx.x;
    if (i < n) deg[i] = rsqrtf(fmaxf(deg[i], 1.0f));
}

// ------------------------- bias init / relu --------------------------------

__global__ void k_init_bias(float* __restrict__ out, const float* __restrict__ b,
                            int F, long long total) {
    long long i = (long long)blockIdx.x * blockDim.x + threadIdx.x;
    if (i < total) out[i] = b[(int)(i % F)];
}

__global__ void k_relu(float* __restrict__ x, long long total) {
    long long i = (long long)blockIdx.x * blockDim.x + threadIdx.x;
    if (i < total) x[i] = fmaxf(x[i], 0.0f);
}

// ------------------------------- WMMA GEMM ---------------------------------
// Out[nrows, COLS] = A[nrows, K] @ W[K, COLS], bf16 operands, f32 accumulate.
// Block: 256 threads = 8 waves; wave w owns rows [blk*128 + w*16, +16).
// W is staged in LDS pre-swizzled into B-fragment order: fragment (s, t, lane)
// = 16 contiguous bf16 (32B) so each lane reads its whole B frag contiguously.
//   B layout (16-bit, 32x16): lanes 0-15 col=lane  K=0..7 & 16..23 (2 per VGPR),
//                             lanes 16-31 col=lane-16 K=8..15 & 24..31.
//   A layout (16-bit, 16x32): lanes 0-15 row M, K = 0..7 & 16..23,
//                             lanes 16-31 row M, K = 8..15 & 24..31.
template<int K, int COLS, int KCHUNK>
__global__ __launch_bounds__(256)
void k_gemm_bf16_wmma(const float* __restrict__ A, const float* __restrict__ W,
                      float* __restrict__ Out, int nrows) {
    constexpr int NT   = COLS / 16;        // N tiles per wave
    constexpr int SS   = KCHUNK / 32;      // k-steps per LDS chunk
    constexpr int LDSE = SS * NT * 512;    // bf16 elems (= SS*NT*32 lanes*16)

    __shared__ __align__(32) __bf16 bl[LDSE];

    const int tid  = threadIdx.x;
    const int wave = tid >> 5;
    const int lane = tid & 31;

    const int rowBase = blockIdx.x * 128 + wave * 16;
    int arow = rowBase + (lane & 15);
    arow = arow < (nrows - 1) ? arow : (nrows - 1);   // clamp, keep EXEC full
    const int hiK = (lane & 16) ? 8 : 0;              // A-frag K sub-offset

    v8f acc[NT];
#pragma unroll
    for (int t = 0; t < NT; ++t) acc[t] = (v8f){0.f,0.f,0.f,0.f,0.f,0.f,0.f,0.f};

    for (int kb = 0; kb < K; kb += KCHUNK) {
        // ---- cooperative load of W chunk, pre-swizzled to fragment order ---
        for (int i = tid; i < LDSE; i += 256) {
            const int s   = i / (NT * 512);
            const int r   = i - s * (NT * 512);
            const int t   = r >> 9;            // / 512
            const int r2  = r & 511;
            const int l   = r2 >> 4;           // lane this elem belongs to
            const int j   = r2 & 15;           // elem within fragment
            const int kk  = s * 32 + ((l & 16) ? 8 : 0) + j + ((j & 8) ? 8 : 0);
            const int col = t * 16 + (l & 15);
            bl[i] = (__bf16)W[(size_t)(kb + kk) * COLS + col];
        }
        __syncthreads();

#pragma unroll
        for (int s = 0; s < SS; ++s) {
            const int k0 = kb + s * 32;
            // ---- A fragment from global (f32 -> bf16) ----------------------
            const float* ap = A + (size_t)arow * K + k0 + hiK;
            v16bf a;
#pragma unroll
            for (int j = 0; j < 8; ++j) a[j]     = (__bf16)ap[j];
#pragma unroll
            for (int j = 0; j < 8; ++j) a[8 + j] = (__bf16)ap[16 + j];

#pragma unroll
            for (int t = 0; t < NT; ++t) {
                const v16bf b =
                    *reinterpret_cast<const v16bf*>(&bl[(((s * NT) + t) * 32 + lane) * 16]);
                acc[t] = __builtin_amdgcn_wmma_f32_16x16x32_bf16(
                    /*neg_a=*/false, a, /*neg_b=*/false, b,
                    /*c_mod=*/(short)0, acc[t],
                    /*reuse_a=*/false, /*reuse_b=*/false);
            }
        }
        __syncthreads();
    }

    // ---- store: C/D layout: VGPR r -> M=r (lanes 0-15) / M=8+r (16-31) -----
    const int rlo   = (lane & 16) ? 8 : 0;
    const int cbase = lane & 15;
#pragma unroll
    for (int t = 0; t < NT; ++t)
#pragma unroll
        for (int r = 0; r < 8; ++r) {
            const int row = rowBase + rlo + r;
            if (row < nrows)
                Out[(size_t)row * COLS + t * 16 + cbase] = acc[t][r];
        }
}

// ------------------------- edge aggregation --------------------------------
// One thread per (edge, 4-float chunk). Edges E..E+N-1 are self-loops.
template<int F>
__global__ void k_aggregate(const float* __restrict__ h, const float* __restrict__ dinv,
                            const int* __restrict__ src, const int* __restrict__ dst,
                            float* __restrict__ out, int nEdges, long long total) {
    constexpr int CH = F / 4;
    long long idx = (long long)blockIdx.x * blockDim.x + threadIdx.x;
    if (idx >= total) return;
    const int e = (int)(idx / CH);
    const int c = (int)(idx - (long long)e * CH);
    int s, d;
    if (e < nEdges) { s = src[e]; d = dst[e]; }
    else            { s = e - nEdges; d = s; }
    const float w = dinv[s] * dinv[d];
    const float4 v = *reinterpret_cast<const float4*>(h + (size_t)s * F + c * 4);
    float* o = out + (size_t)d * F + c * 4;
    atomicAdd(o + 0, v.x * w);
    atomicAdd(o + 1, v.y * w);
    atomicAdd(o + 2, v.z * w);
    atomicAdd(o + 3, v.w * w);
}

// ------------------------------- launcher ----------------------------------

extern "C" void kernel_launch(void* const* d_in, const int* in_sizes, int n_in,
                              void* d_out, int out_size, void* d_ws, size_t ws_size,
                              hipStream_t stream) {
    const float* x   = (const float*)d_in[0];
    const int*   ei  = (const int*)  d_in[1];
    const float* W1  = (const float*)d_in[2];
    const float* b1  = (const float*)d_in[3];
    const float* W2  = (const float*)d_in[4];
    const float* b2  = (const float*)d_in[5];
    float* out = (float*)d_out;

    const int N = in_sizes[0] / GCN_DIN;
    const int E = in_sizes[1] / 2;
    const int* src = ei;
    const int* dst = ei + E;

    // workspace carve-up
    char* ws = (char*)d_ws;
    float* dinv = (float*)ws;                 ws += ((size_t)N * 4 + 255) & ~(size_t)255;
    float* h1   = (float*)ws;                 ws += (size_t)N * GCN_DH * 4;
    float* agg1 = (float*)ws;                 ws += (size_t)N * GCN_DH * 4;
    float* h2   = (float*)ws;

    const int B = 256;

    // ---- degree -> dinv ----
    k_init_deg   <<<(N + B - 1) / B, B, 0, stream>>>(dinv, N);
    k_deg_hist   <<<(E + B - 1) / B, B, 0, stream>>>(dst, dinv, E);
    k_deg_to_dinv<<<(N + B - 1) / B, B, 0, stream>>>(dinv, N);

    // ---- layer 1 ----
    k_gemm_bf16_wmma<GCN_DIN, GCN_DH, 64>
        <<<(N + 127) / 128, 256, 0, stream>>>(x, W1, h1, N);

    long long tot1 = (long long)N * GCN_DH;
    k_init_bias<<<(unsigned)((tot1 + B - 1) / B), B, 0, stream>>>(agg1, b1, GCN_DH, tot1);

    long long work1 = (long long)(E + N) * (GCN_DH / 4);
    k_aggregate<GCN_DH>
        <<<(unsigned)((work1 + B - 1) / B), B, 0, stream>>>(h1, dinv, src, dst, agg1, E, work1);

    k_relu<<<(unsigned)((tot1 + B - 1) / B), B, 0, stream>>>(agg1, tot1);

    // ---- layer 2 ----
    k_gemm_bf16_wmma<GCN_DH, GCN_DOUT, 128>
        <<<(N + 127) / 128, 256, 0, stream>>>(agg1, W2, h2, N);

    long long tot2 = (long long)N * GCN_DOUT;
    k_init_bias<<<(unsigned)((tot2 + B - 1) / B), B, 0, stream>>>(out, b2, GCN_DOUT, tot2);

    long long work2 = (long long)(E + N) * (GCN_DOUT / 4);
    k_aggregate<GCN_DOUT>
        <<<(unsigned)((work2 + B - 1) / B), B, 0, stream>>>(h2, dinv, src, dst, out, E, work2);
}
